// EdgeClassifierGAT3Layer_1949915152971
// MI455X (gfx1250) — matrix operand order repaired and checked by reading, not compiled, not run on hardware
//
#include <hip/hip_runtime.h>
#include <hip/hip_bf16.h>

// ---------------------------------------------------------------------------
// Types for CDNA5 WMMA (wave32, v_wmma_f32_16x16x32_bf16)
// ---------------------------------------------------------------------------
typedef __bf16 bf16_t;
typedef bf16_t v16bf __attribute__((ext_vector_type(16)));
typedef bf16_t v8bf  __attribute__((ext_vector_type(8)));
typedef bf16_t v4bf  __attribute__((ext_vector_type(4)));
typedef bf16_t v2bf  __attribute__((ext_vector_type(2)));
typedef float  v8f   __attribute__((ext_vector_type(8)));
typedef int    v4i_t __attribute__((ext_vector_type(4)));
typedef int    v8i_t __attribute__((ext_vector_type(8)));

static __device__ inline bf16_t f2bf(float f) { return (bf16_t)f; }

static __device__ inline v8f zero8() {
    v8f z;
#pragma unroll
    for (int i = 0; i < 8; ++i) z[i] = 0.0f;
    return z;
}

// A-operand 16x32 bf16 from LDS, row-major, stride in elements (stride*2B and
// base must be 16B aligned). Two contiguous 16B runs per lane -> ds_load_b128.
static __device__ inline v16bf load_frag_A(const bf16_t* base, int stride, int lane) {
    int row = lane & 15;
    int kb  = (lane >> 4) << 3;   // 0 or 8
    const bf16_t* p = base + row * stride + kb;
    v8bf lo = *(const v8bf*)p;
    v8bf hi = *(const v8bf*)(p + 16);
    return __builtin_shufflevector(lo, hi, 0, 1, 2, 3, 4, 5, 6, 7,
                                   8, 9, 10, 11, 12, 13, 14, 15);
}

// B-operand 32x16 bf16 from LDS in fragment-major layout: BsT[col][k], k 0..31
// contiguous per column. lanes 0-15: col=lane, K 0-7,16-23; lanes 16-31: K 8-15,24-31.
static __device__ inline v16bf load_frag_Bt(const bf16_t* baseT, int lane) {
    int col = lane & 15;
    int kh  = (lane >> 4) << 3;   // 0 or 8
    const bf16_t* p = baseT + col * 32 + kh;
    v8bf lo = *(const v8bf*)p;
    v8bf hi = *(const v8bf*)(p + 16);
    return __builtin_shufflevector(lo, hi, 0, 1, 2, 3, 4, 5, 6, 7,
                                   8, 9, 10, 11, 12, 13, 14, 15);
}

static __device__ inline v8f wmma_bf16(v16bf a, v16bf b, v8f c) {
    return __builtin_amdgcn_wmma_f32_16x16x32_bf16(false, a, false, b, (short)0, c,
                                                   false, false);
}

// Order-preserving float<->uint encoding for atomic max
static __device__ inline unsigned fenc(float f) {
    unsigned u = __float_as_uint(f);
    return (u & 0x80000000u) ? ~u : (u | 0x80000000u);
}
static __device__ inline float fdec(unsigned k) {
    unsigned u = (k & 0x80000000u) ? (k & 0x7FFFFFFFu) : ~k;
    return __uint_as_float(u);
}

// ---------------------------------------------------------------------------
// Tensor Data Mover: 2D bf16 tile (tile_d0 x tile_d1) -> LDS (packed rows)
// ---------------------------------------------------------------------------
static __device__ inline void tdm_load_2d_bf16(unsigned lds_addr, const bf16_t* gptr,
                                               unsigned tensor_d0, unsigned tensor_d1,
                                               unsigned tile_d0, unsigned tile_d1,
                                               unsigned stride_d0) {
    unsigned long long ga = (unsigned long long)(uintptr_t)gptr;
    v4i_t g0;
    g0[0] = 1;                                        // count=1, user mode
    g0[1] = (int)lds_addr;                            // lds_addr (bytes)
    g0[2] = (int)(unsigned)(ga & 0xffffffffu);        // global_addr lo
    g0[3] = (int)((unsigned)((ga >> 32) & 0x1ffffffu) | (2u << 30)); // hi | type=2
    v8i_t g1;
    g1[0] = (int)(1u << 16);                          // data_size=1 (2 bytes)
    g1[1] = (int)((tensor_d0 & 0xffffu) << 16);       // tensor_dim0[15:0]
    g1[2] = (int)((tensor_d0 >> 16) | ((tensor_d1 & 0xffffu) << 16));
    g1[3] = (int)((tensor_d1 >> 16) | (tile_d0 << 16));
    g1[4] = (int)tile_d1;                             // tile_dim1 (tile_dim2=0)
    g1[5] = (int)stride_d0;                           // tensor_dim0_stride lo32
    g1[6] = 0;
    g1[7] = 0;
    asm volatile("tensor_load_to_lds %0, %1" :: "s"(g0), "s"(g1) : "memory");
}

// ---------------------------------------------------------------------------
// Weight pre-conversion with padding: WT[n*Kp + k] = bf16(W[k*N + n]) or 0
// ---------------------------------------------------------------------------
__global__ __launch_bounds__(256) void convert_wT_pad(const float* __restrict__ W,
                                                      bf16_t* __restrict__ WT,
                                                      int K, int N, int Kp, int Np) {
    int i = blockIdx.x * 256 + threadIdx.x;
    if (i >= Kp * Np) return;
    int n = i / Kp, k = i - n * Kp;
    WT[i] = f2bf((k < K && n < N) ? W[(size_t)k * N + n] : 0.f);
}

// ---------------------------------------------------------------------------
// Mean of edge_attr over edges (self-loop fill_value='mean')
// ---------------------------------------------------------------------------
__global__ __launch_bounds__(512) void ea_mean_kernel(const float* __restrict__ ea,
                                                      float* __restrict__ mean, int E) {
    __shared__ float s[512];
    int t = threadIdx.x;
    int col = t & 15, grp = t >> 4;
    float acc = 0.f;
    for (int r = grp; r < E; r += 32) acc += ea[(size_t)r * 16 + col];
    s[t] = acc;
    __syncthreads();
    for (int off = 256; off >= 16; off >>= 1) {
        if (t < off) s[t] += s[t + off];
        __syncthreads();
    }
    if (t < 16) mean[t] = s[t] / (float)E;
}

// ---------------------------------------------------------------------------
// WMMA GEMM: C[M,N] = A[M,K] @ W[K,N] + bias.  W pre-transposed bf16 BT[N][K];
// B-tiles streamed by double-buffered Tensor Data Mover (overlaps compute).
// K % 32 == 0, N % 64 == 0. Block = 128 threads (4 waves), tile 64x64.
// ---------------------------------------------------------------------------
__global__ __launch_bounds__(128) void gemm_wmma(const float* __restrict__ Ag,
                                                 const bf16_t* __restrict__ BT,
                                                 const float* __restrict__ bias,
                                                 float* __restrict__ C,
                                                 int M, int N, int K) {
    __shared__ bf16_t As[64 * 32];       // A tile, row-major [64][32]
    __shared__ bf16_t BsT[2][64 * 32];   // B tiles, fragment-major, ping-pong
    int t = threadIdx.x, lane = t & 31, wave = t >> 5;
    int m0 = blockIdx.x * 64;
    int n0 = blockIdx.y * 64;
    unsigned boff0 = (unsigned)(uintptr_t)(void*)&BsT[0][0];
    unsigned boff1 = (unsigned)(uintptr_t)(void*)&BsT[1][0];
    bool fullM = (m0 + 64) <= M;
    int nk = K >> 5;

    v8f acc[4];
#pragma unroll
    for (int i = 0; i < 4; ++i) acc[i] = zero8();

    if (wave == 0)   // prologue: DMA first B tile
        tdm_load_2d_bf16(boff0, BT + (size_t)n0 * K, (unsigned)K, (unsigned)N,
                         32u, 64u, (unsigned)K);

    for (int i = 0; i < nk; ++i) {
        int k0 = i << 5;
        // stage A tile (fp32 -> bf16)
        if (fullM) {
            for (int idx = t; idx < 512; idx += 128) {
                int r = idx >> 3, k4 = (idx & 7) << 2;
                float4 f4 = *(const float4*)&Ag[(size_t)(m0 + r) * K + k0 + k4];
                v4bf pk;
                pk[0] = f2bf(f4.x); pk[1] = f2bf(f4.y);
                pk[2] = f2bf(f4.z); pk[3] = f2bf(f4.w);
                *(v4bf*)&As[r * 32 + k4] = pk;
            }
        } else {
            for (int idx = t; idx < 64 * 16; idx += 128) {
                int r = idx >> 4, k2 = idx & 15;
                int gr = m0 + r;
                float2 t2; t2.x = 0.f; t2.y = 0.f;
                if (gr < M) t2 = *(const float2*)&Ag[(size_t)gr * K + k0 + 2 * k2];
                v2bf pk; pk[0] = f2bf(t2.x); pk[1] = f2bf(t2.y);
                *(v2bf*)&As[r * 32 + 2 * k2] = pk;
            }
        }
        if (k0 + 32 < K) {   // prefetch next A k-tile
            int pr = m0 + (t & 63);
            if (pr < M) __builtin_prefetch(&Ag[(size_t)pr * K + k0 + 32], 0, 1);
        }
        // DMA next B tile into the other buffer, wait for current one
        if (wave == 0) {
            if (i + 1 < nk) {
                tdm_load_2d_bf16((i & 1) ? boff0 : boff1,
                                 BT + (size_t)n0 * K + k0 + 32,
                                 (unsigned)K, (unsigned)N, 32u, 64u, (unsigned)K);
                __builtin_amdgcn_s_wait_tensorcnt(1);
            } else {
                __builtin_amdgcn_s_wait_tensorcnt(0);
            }
        }
        __syncthreads();
        const bf16_t* bb = &BsT[i & 1][0];
        v16bf a = load_frag_A(&As[wave * 16 * 32], 32, lane);
#pragma unroll
        for (int nt = 0; nt < 4; ++nt)
            acc[nt] = wmma_bf16(a, load_frag_Bt(bb + nt * 16 * 32, lane), acc[nt]);
        __syncthreads();
    }

    int n = lane & 15, mb = (lane >> 4) << 3;
#pragma unroll
    for (int nt = 0; nt < 4; ++nt) {
        int col = n0 + nt * 16 + n;
        float bv = bias ? bias[col] : 0.f;
#pragma unroll
        for (int r = 0; r < 8; ++r) {
            int row = m0 + wave * 16 + mb + r;
            if (row < M) C[(size_t)row * N + col] = acc[nt][r] + bv;
        }
    }
}

// ---------------------------------------------------------------------------
// Edge attention logits (fused ea@We) + atomic max per (dst, head)
// ---------------------------------------------------------------------------
template <int HC, int H>
__global__ __launch_bounds__(256) void edge_attn_logits(
    const float* __restrict__ xl, const float* __restrict__ xr,
    const float* __restrict__ ea, const float* __restrict__ eamean,
    const long long* __restrict__ srcI, const long long* __restrict__ dstI,
    const float* __restrict__ We, const float* __restrict__ att,
    float* __restrict__ alpha, unsigned* __restrict__ amaxk, int E, int ET) {
    int wv = blockIdx.x * 8 + (threadIdx.x >> 5);
    if (wv >= ET) return;
    int lane = threadIdx.x & 31;
    int e = wv, s, d;
    const float* eap;
    if (e < E) { s = (int)srcI[e]; d = (int)dstI[e]; eap = ea + (size_t)e * 16; }
    else       { s = d = e - E;    eap = eamean; }

    float er[16];
#pragma unroll
    for (int k = 0; k < 16; ++k) er[k] = eap[k];

    const int CPL = HC / 32;
    const int LP  = 32 / H;
    const float* xls = xl + (size_t)s * HC;
    const float* xrd = xr + (size_t)d * HC;
    float p = 0.f;
#pragma unroll
    for (int i = 0; i < CPL; ++i) {
        int c = lane * CPL + i;
        float m = xls[c] + xrd[c];
#pragma unroll
        for (int k = 0; k < 16; ++k) m += er[k] * We[k * HC + c];
        float sv = m > 0.f ? m : 0.2f * m;
        p += sv * att[c];
    }
#pragma unroll
    for (int off = 1; off < LP; off <<= 1) p += __shfl_xor(p, off, 32);
    if ((lane & (LP - 1)) == 0) {
        int h = lane / LP;
        alpha[(size_t)e * H + h] = p;
        atomicMax(&amaxk[d * H + h], fenc(p));
    }
}

// exp(alpha - max) and denominator accumulation
__global__ __launch_bounds__(256) void edge_softmax_num(
    float* __restrict__ alpha, const unsigned* __restrict__ amaxk,
    float* __restrict__ den, const long long* __restrict__ dstI,
    int E, int ET, int H) {
    int i = blockIdx.x * blockDim.x + threadIdx.x;
    if (i >= ET * H) return;
    int e = i / H, h = i - e * H;
    int d = e < E ? (int)dstI[e] : (e - E);
    float ex = expf(alpha[i] - fdec(amaxk[d * H + h]));
    alpha[i] = ex;
    atomicAdd(&den[d * H + h], ex);
}

// weighted scatter aggregation: agg[dst] += xl[src] * w
template <int HC, int H>
__global__ __launch_bounds__(256) void edge_aggregate(
    const float* __restrict__ xl, const float* __restrict__ alpha,
    const float* __restrict__ den, const long long* __restrict__ srcI,
    const long long* __restrict__ dstI, float* __restrict__ agg, int E, int ET) {
    int wv = blockIdx.x * 8 + (threadIdx.x >> 5);
    if (wv >= ET) return;
    int lane = threadIdx.x & 31;
    int e = wv, s, d;
    if (e < E) { s = (int)srcI[e]; d = (int)dstI[e]; }
    else       { s = d = e - E; }
    const int CPL = HC / 32, LP = 32 / H;
    int h = lane / LP;
    float w = alpha[(size_t)e * H + h] / den[d * H + h];
    const float* xs = xl + (size_t)s * HC;
    float* ag = agg + (size_t)d * HC;
#pragma unroll
    for (int i = 0; i < CPL; ++i) {
        int c = lane * CPL + i;
        atomicAdd(&ag[c], xs[c] * w);
    }
}

// out = act(in + bias[col])
__global__ __launch_bounds__(256) void bias_act(const float* __restrict__ in,
                                                const float* __restrict__ bias,
                                                float* __restrict__ out,
                                                int n, int W, int relu) {
    int i = blockIdx.x * blockDim.x + threadIdx.x;
    if (i >= n) return;
    float v = in[i] + bias[i % W];
    out[i] = (relu && v < 0.f) ? 0.f : v;
}

// ---------------------------------------------------------------------------
// Fused edge MLP: [h[src], h[dst], ea] (272, padded 288) ->128 ->128 ->2
// All weight tiles (pre-converted bf16, transposed, padded) streamed by
// double-buffered TDM; per-wave 16-edge tiles; all GEMMs via WMMA bf16.
// ---------------------------------------------------------------------------
#define MK1 288
__global__ __launch_bounds__(128) void mlp_fused(
    const float* __restrict__ h, const long long* __restrict__ srcI,
    const long long* __restrict__ dstI, const float* __restrict__ ea,
    const bf16_t* __restrict__ W1T, const float* __restrict__ b1,
    const bf16_t* __restrict__ W2T, const float* __restrict__ b2,
    const bf16_t* __restrict__ W3T, const float* __restrict__ b3,
    float* __restrict__ out, int E) {
    __shared__ bf16_t Ash[4][16 * MK1];   // per-wave A tiles
    __shared__ bf16_t Wsh[2][128 * 32];   // weight tiles, fragment-major, ping-pong
    int t = threadIdx.x, lane = t & 31, wave = t >> 5;
    int tileBase = (blockIdx.x * 4 + wave) * 16;
    bf16_t* A = &Ash[wave][0];
    unsigned woff0 = (unsigned)(uintptr_t)(void*)&Wsh[0][0];
    unsigned woff1 = (unsigned)(uintptr_t)(void*)&Wsh[1][0];

    // build input tile [16, 288]: pair-packed bf16 stores
    for (int idx = lane; idx < 16 * (MK1 / 2); idx += 32) {
        int r = idx / (MK1 / 2), c2 = idx % (MK1 / 2);
        int c = 2 * c2;
        int e = tileBase + r;
        float2 t2; t2.x = 0.f; t2.y = 0.f;
        if (e < E) {
            if (c < 128)      t2 = *(const float2*)&h[(size_t)((int)srcI[e]) * 128 + c];
            else if (c < 256) t2 = *(const float2*)&h[(size_t)((int)dstI[e]) * 128 + (c - 128)];
            else if (c < 272) t2 = *(const float2*)&ea[(size_t)e * 16 + (c - 256)];
        }
        v2bf pk; pk[0] = f2bf(t2.x); pk[1] = f2bf(t2.y);
        *(v2bf*)&A[r * MK1 + c] = pk;
    }
    __syncthreads();

    v8f acc[8];
    int n = lane & 15, mb = (lane >> 4) << 3;

    // ---- layer 1: K=288 (272 real), N=128 ----
#pragma unroll
    for (int nt = 0; nt < 8; ++nt) acc[nt] = zero8();
    if (wave == 0)
        tdm_load_2d_bf16(woff0, W1T, 288u, 128u, 32u, 128u, 288u);
    for (int i = 0; i < 9; ++i) {
        int k0 = i << 5;
        if (wave == 0) {
            if (i + 1 < 9) {
                tdm_load_2d_bf16((i & 1) ? woff0 : woff1, W1T + (i + 1) * 32,
                                 288u, 128u, 32u, 128u, 288u);
                __builtin_amdgcn_s_wait_tensorcnt(1);
            } else {
                __builtin_amdgcn_s_wait_tensorcnt(0);
            }
        }
        __syncthreads();
        const bf16_t* wb = &Wsh[i & 1][0];
        v16bf a = load_frag_A(A + k0, MK1, lane);
#pragma unroll
        for (int nt = 0; nt < 8; ++nt)
            acc[nt] = wmma_bf16(a, load_frag_Bt(wb + nt * 16 * 32, lane), acc[nt]);
        __syncthreads();
    }
#pragma unroll
    for (int nt = 0; nt < 8; ++nt) {
        int col = nt * 16 + n;
        float bv = b1[col];
#pragma unroll
        for (int r = 0; r < 8; ++r) {
            float v = acc[nt][r] + bv;
            A[(mb + r) * MK1 + col] = f2bf(v > 0.f ? v : 0.f);
        }
    }
    __syncthreads();

    // ---- layer 2: K=128, N=128 ----
#pragma unroll
    for (int nt = 0; nt < 8; ++nt) acc[nt] = zero8();
    if (wave == 0)
        tdm_load_2d_bf16(woff0, W2T, 128u, 128u, 32u, 128u, 128u);
    for (int i = 0; i < 4; ++i) {
        int k0 = i << 5;
        if (wave == 0) {
            if (i + 1 < 4) {
                tdm_load_2d_bf16((i & 1) ? woff0 : woff1, W2T + (i + 1) * 32,
                                 128u, 128u, 32u, 128u, 128u);
                __builtin_amdgcn_s_wait_tensorcnt(1);
            } else {
                __builtin_amdgcn_s_wait_tensorcnt(0);
            }
        }
        __syncthreads();
        const bf16_t* wb = &Wsh[i & 1][0];
        v16bf a = load_frag_A(A + k0, MK1, lane);
#pragma unroll
        for (int nt = 0; nt < 8; ++nt)
            acc[nt] = wmma_bf16(a, load_frag_Bt(wb + nt * 16 * 32, lane), acc[nt]);
        __syncthreads();
    }
#pragma unroll
    for (int nt = 0; nt < 8; ++nt) {
        int col = nt * 16 + n;
        float bv = b2[col];
#pragma unroll
        for (int r = 0; r < 8; ++r) {
            float v = acc[nt][r] + bv;
            A[(mb + r) * MK1 + col] = f2bf(v > 0.f ? v : 0.f);
        }
    }
    __syncthreads();

    // ---- layer 3: K=128, N=2 (padded to 16 in W3T) ----
    v8f acc3 = zero8();
    if (wave == 0)
        tdm_load_2d_bf16(woff0, W3T, 128u, 16u, 32u, 16u, 128u);
    for (int i = 0; i < 4; ++i) {
        int k0 = i << 5;
        if (wave == 0) {
            if (i + 1 < 4) {
                tdm_load_2d_bf16((i & 1) ? woff0 : woff1, W3T + (i + 1) * 32,
                                 128u, 16u, 32u, 16u, 128u);
                __builtin_amdgcn_s_wait_tensorcnt(1);
            } else {
                __builtin_amdgcn_s_wait_tensorcnt(0);
            }
        }
        __syncthreads();
        const bf16_t* wb = &Wsh[i & 1][0];
        v16bf a = load_frag_A(A + k0, MK1, lane);
        acc3 = wmma_bf16(a, load_frag_Bt(wb, lane), acc3);
        __syncthreads();
    }
    if (n < 2) {
#pragma unroll
        for (int r = 0; r < 8; ++r) {
            int e = tileBase + mb + r;
            if (e < E) out[(size_t)e * 2 + n] = acc3[r] + b3[n];
        }
    }
}

// ---------------------------------------------------------------------------
// Host launcher
// ---------------------------------------------------------------------------
static inline int cdiv(int a, int b) { return (a + b - 1) / b; }

extern "C" void kernel_launch(void* const* d_in, const int* in_sizes, int n_in,
                              void* d_out, int out_size, void* d_ws, size_t ws_size,
                              hipStream_t stream) {
    const float*     x    = (const float*)d_in[0];
    const long long* ei   = (const long long*)d_in[1];
    const float*     ea   = (const float*)d_in[2];

    const float* p[30];
    for (int i = 3; i < 30 && i < n_in; ++i) p[i] = (const float*)d_in[i];

    const int IN_C = 128, EDGE_DIM = 16;
    int Nn = in_sizes[0] / IN_C;
    int E  = in_sizes[2] / EDGE_DIM;
    int ET = E + Nn;
    const long long* srcI = ei;
    const long long* dstI = ei + E;

    // workspace carve-up
    char* w = (char*)d_ws;
    auto take = [&](size_t bytes) { char* r = w; w += (bytes + 255) & ~(size_t)255; return r; };
    size_t NHC = (size_t)Nn * 512;
    float*    eamean = (float*)take(64 * sizeof(float));
    float*    xl     = (float*)take(NHC * sizeof(float));
    float*    xr     = (float*)take(NHC * sizeof(float));
    float*    agg    = (float*)take(NHC * sizeof(float));
    float*    hA     = (float*)take(NHC * sizeof(float));
    float*    hB     = (float*)take(NHC * sizeof(float));
    float*    alpha  = (float*)take((size_t)ET * 4 * sizeof(float));
    unsigned* amaxk  = (unsigned*)take((size_t)Nn * 4 * sizeof(unsigned));
    float*    den    = (float*)take((size_t)Nn * 4 * sizeof(float));
    // bf16 transposed (padded) weights for TDM-fed GEMMs
    bf16_t* WT1l = (bf16_t*)take(128 * 512 * sizeof(bf16_t));
    bf16_t* WT1r = (bf16_t*)take(128 * 512 * sizeof(bf16_t));
    bf16_t* WT2l = (bf16_t*)take(512 * 512 * sizeof(bf16_t));
    bf16_t* WT2r = (bf16_t*)take(512 * 512 * sizeof(bf16_t));
    bf16_t* WT3l = (bf16_t*)take(512 * 128 * sizeof(bf16_t));
    bf16_t* WT3r = (bf16_t*)take(512 * 128 * sizeof(bf16_t));
    bf16_t* W1T  = (bf16_t*)take(288 * 128 * sizeof(bf16_t));
    bf16_t* W2T  = (bf16_t*)take(128 * 128 * sizeof(bf16_t));
    bf16_t* W3T  = (bf16_t*)take(128 * 16 * sizeof(bf16_t));

    convert_wT_pad<<<cdiv(128 * 512, 256), 256, 0, stream>>>(p[3],  WT1l, 128, 512, 128, 512);
    convert_wT_pad<<<cdiv(128 * 512, 256), 256, 0, stream>>>(p[5],  WT1r, 128, 512, 128, 512);
    convert_wT_pad<<<cdiv(512 * 512, 256), 256, 0, stream>>>(p[10], WT2l, 512, 512, 512, 512);
    convert_wT_pad<<<cdiv(512 * 512, 256), 256, 0, stream>>>(p[12], WT2r, 512, 512, 512, 512);
    convert_wT_pad<<<cdiv(512 * 128, 256), 256, 0, stream>>>(p[17], WT3l, 512, 128, 512, 128);
    convert_wT_pad<<<cdiv(512 * 128, 256), 256, 0, stream>>>(p[19], WT3r, 512, 128, 512, 128);
    convert_wT_pad<<<cdiv(288 * 128, 256), 256, 0, stream>>>(p[24], W1T, 272, 128, 288, 128);
    convert_wT_pad<<<cdiv(128 * 128, 256), 256, 0, stream>>>(p[26], W2T, 128, 128, 128, 128);
    convert_wT_pad<<<cdiv(128 * 16, 256), 256, 0, stream>>>(p[28], W3T, 128, 2, 128, 16);

    ea_mean_kernel<<<1, 512, 0, stream>>>(ea, eamean, E);

    dim3 blk(128);
    int gM = cdiv(Nn, 64);
    int eg = cdiv(ET, 8);

    // ---- GAT layer 1: in 128, H=4, C=128 (HC=512) ----
    gemm_wmma<<<dim3(gM, 8), blk, 0, stream>>>(x, WT1l, p[4], xl, Nn, 512, 128);
    gemm_wmma<<<dim3(gM, 8), blk, 0, stream>>>(x, WT1r, p[6], xr, Nn, 512, 128);
    hipMemsetAsync(amaxk, 0, (size_t)Nn * 4 * sizeof(unsigned), stream);
    hipMemsetAsync(den, 0, (size_t)Nn * 4 * sizeof(float), stream);
    hipMemsetAsync(agg, 0, NHC * sizeof(float), stream);
    edge_attn_logits<512, 4><<<eg, 256, 0, stream>>>(xl, xr, ea, eamean, srcI, dstI,
                                                     p[7], p[8], alpha, amaxk, E, ET);
    edge_softmax_num<<<cdiv(ET * 4, 256), 256, 0, stream>>>(alpha, amaxk, den, dstI, E, ET, 4);
    edge_aggregate<512, 4><<<eg, 256, 0, stream>>>(xl, alpha, den, srcI, dstI, agg, E, ET);
    bias_act<<<cdiv((int)NHC, 256), 256, 0, stream>>>(agg, p[9], hA, (int)NHC, 512, 1);

    // ---- GAT layer 2: in 512, H=4, C=128 ----
    gemm_wmma<<<dim3(gM, 8), blk, 0, stream>>>(hA, WT2l, p[11], xl, Nn, 512, 512);
    gemm_wmma<<<dim3(gM, 8), blk, 0, stream>>>(hA, WT2r, p[13], xr, Nn, 512, 512);
    hipMemsetAsync(amaxk, 0, (size_t)Nn * 4 * sizeof(unsigned), stream);
    hipMemsetAsync(den, 0, (size_t)Nn * 4 * sizeof(float), stream);
    hipMemsetAsync(agg, 0, NHC * sizeof(float), stream);
    edge_attn_logits<512, 4><<<eg, 256, 0, stream>>>(xl, xr, ea, eamean, srcI, dstI,
                                                     p[14], p[15], alpha, amaxk, E, ET);
    edge_softmax_num<<<cdiv(ET * 4, 256), 256, 0, stream>>>(alpha, amaxk, den, dstI, E, ET, 4);
    edge_aggregate<512, 4><<<eg, 256, 0, stream>>>(xl, alpha, den, srcI, dstI, agg, E, ET);
    bias_act<<<cdiv((int)NHC, 256), 256, 0, stream>>>(agg, p[16], hB, (int)NHC, 512, 1);

    // ---- GAT layer 3: in 512, H=1, C=128 (concat=False, mean over 1 head) ----
    gemm_wmma<<<dim3(gM, 2), blk, 0, stream>>>(hB, WT3l, p[18], xl, Nn, 128, 512);
    gemm_wmma<<<dim3(gM, 2), blk, 0, stream>>>(hB, WT3r, p[20], xr, Nn, 128, 512);
    hipMemsetAsync(amaxk, 0, (size_t)Nn * sizeof(unsigned), stream);
    hipMemsetAsync(den, 0, (size_t)Nn * sizeof(float), stream);
    hipMemsetAsync(agg, 0, (size_t)Nn * 128 * sizeof(float), stream);
    edge_attn_logits<128, 1><<<eg, 256, 0, stream>>>(xl, xr, ea, eamean, srcI, dstI,
                                                     p[21], p[22], alpha, amaxk, E, ET);
    edge_softmax_num<<<cdiv(ET, 256), 256, 0, stream>>>(alpha, amaxk, den, dstI, E, ET, 1);
    edge_aggregate<128, 1><<<eg, 256, 0, stream>>>(xl, alpha, den, srcI, dstI, agg, E, ET);
    bias_act<<<cdiv(Nn * 128, 256), 256, 0, stream>>>(agg, p[23], hA, Nn * 128, 128, 1);

    // ---- fused edge MLP ----
    mlp_fused<<<cdiv(E, 64), blk, 0, stream>>>(hA, srcI, dstI, ea,
                                               W1T, p[25], W2T, p[27], W3T, p[29],
                                               (float*)d_out, E);
}
